// MambaModel_56238301774003
// MI455X (gfx1250) — compile-verified, matrix operand order checked by reading, not run on hardware
//
#include <hip/hip_runtime.h>
#include <hip/hip_bf16.h>
#include <math.h>

// ---------------- problem constants ----------------
#define B_      8
#define T_      4096
#define D_      128
#define N_      128
#define HID_    256
#define P_      128
#define NBLK_   4
#define CHUNK_  128
#define NC_     (T_ / CHUNK_)        // 32 chunks per sequence
#define TOK_    (B_ * T_ / 2)        // 16384 tokens per head
#define TOKP_   ((size_t)TOK_ * P_)  // 2,097,152 elements per output tensor

typedef __attribute__((ext_vector_type(16))) __bf16 v16bf;
typedef __attribute__((ext_vector_type(8)))  __bf16 v8bf;
typedef __attribute__((ext_vector_type(4)))  __bf16 v4bf;
typedef __attribute__((ext_vector_type(8)))  float  v8f;

// LDS row strides (bf16 elems), padded so row stride % 64 dwords == 4 -> no
// 16-way bank conflicts when 16 lanes read 16 consecutive rows.
#define XROWS_  18                   // conv tile rows: t0-1 .. t0+16
#define XSTR_   136                  // 128 + 8  (272 B)
#define ASTR_   264                  // 256 + 8  (528 B), max Kdim = 256

__device__ __forceinline__ v16bf frag_from_lds(const __bf16* p) {
    // lane's 16 elements = two contiguous 8-elem (16 B) runs, 32 B apart
    v8bf lo = *(const v8bf*)(p);
    v8bf hi = *(const v8bf*)(p + 16);
    return __builtin_shufflevector(lo, hi, 0,1,2,3,4,5,6,7,8,9,10,11,12,13,14,15);
}

// ---------------------------------------------------------------------------
// 0) Weight repack: W [Kdim x Nn] f32 row-major -> bf16 in WMMA B-fragment
//    order: out[((kk*ntiles + nt)*32 + lane)*16 + e], so each lane's 16
//    elements of a fragment are one contiguous 32 B run (two b128 loads).
// ---------------------------------------------------------------------------
__global__ void repack_w_kernel(const float* __restrict__ W, __bf16* __restrict__ out,
                                int Kdim, int Nn) {
    int q = blockIdx.x * blockDim.x + threadIdx.x;     // one thread per (frag,lane)
    int ntiles = Nn >> 4;
    int total = (Kdim >> 5) * ntiles * 32;
    if (q >= total) return;
    int lane = q & 31;
    int f    = q >> 5;
    int nt   = f % ntiles;
    int kk   = f / ntiles;
    int half = lane >> 4, lr = lane & 15;
    v16bf v;
    #pragma unroll
    for (int e = 0; e < 16; ++e) {
        int k = ((e >> 3) << 4) + (half << 3) + (e & 7);
        v[e] = (__bf16)W[(size_t)(kk * 32 + k) * Nn + nt * 16 + lr];
    }
    *(v16bf*)(out + (size_t)q * 16) = v;
}

// ---------------------------------------------------------------------------
// 1) Per-block coefficient tables: Abar = exp(delta*A_sel), Bbar = delta*B_sel,
//    C_sel.  Selection by t parity (type_ids == t % 2; is_coarse == odd t).
// ---------------------------------------------------------------------------
__global__ void coef_kernel(const float* __restrict__ A1, const float* __restrict__ A2,
                            const float* __restrict__ B1, const float* __restrict__ B2,
                            const float* __restrict__ C1, const float* __restrict__ C2,
                            const float* __restrict__ delta,
                            float* __restrict__ Abar, float* __restrict__ Bbar,
                            float* __restrict__ Cc) {
    int i = blockIdx.x * blockDim.x + threadIdx.x;
    if (i >= T_ * N_) return;
    int t = i / N_;
    bool coarse = (t & 1) != 0;          // type_ids == 1
    float a  = coarse ? A1[i] : A2[i];
    float bm = coarse ? B1[i] : B2[i];
    float c  = coarse ? C1[i] : C2[i];
    float dl = delta[i];
    Abar[i] = __expf(dl * a);
    Bbar[i] = dl * bm;
    Cc[i]   = c;
}

// ---------------------------------------------------------------------------
// 2) Dual conv1d (K=3, SAME) + bias + ReLU for gate & SSM branches.
//    Block = 256 threads = 8 waves: one 16-row t-tile, waves cover all 8
//    n-tiles.  The 18-row x tile is staged ONCE in LDS as bf16; A fragments
//    are two ds_load_b128, W fragments are one v16bf load from the repacked
//    bf16 weights.  3 taps x 4 K-chunks x 2 outputs = 24 wmma/wave.
// ---------------------------------------------------------------------------
__global__ __launch_bounds__(256)
void conv_gz_kernel(const float* __restrict__ x,        // [B,T,D]
                    const __bf16* __restrict__ wgrep, const float* __restrict__ biasG,
                    const __bf16* __restrict__ wzrep, const float* __restrict__ biasZ,
                    float* __restrict__ g, float* __restrict__ z) {
    __shared__ __bf16 sx[XROWS_ * XSTR_];

    const int lane = threadIdx.x & 31;
    const int wave = threadIdx.x >> 5;
    const int half = lane >> 4;
    const int lr   = lane & 15;
    const int t0   = blockIdx.x * 16;
    const int b    = blockIdx.y;
    const int n0   = wave * 16;

    const float* xb = x + (size_t)b * T_ * D_;

    // ---- cooperative stage: 18 rows x 32 float4 = 576 chunks over 256 thr ----
    for (int q = threadIdx.x; q < XROWS_ * 32; q += 256) {
        int row = q >> 5;                 // 0..17  -> t = t0 + row - 1
        int c4  = q & 31;                 // float4 index within row
        int t   = t0 + row - 1;
        float4 v = make_float4(0.f, 0.f, 0.f, 0.f);
        if (t >= 0 && t < T_) v = *(const float4*)(xb + (size_t)t * D_ + c4 * 4);
        v4bf w;
        w[0] = (__bf16)v.x; w[1] = (__bf16)v.y; w[2] = (__bf16)v.z; w[3] = (__bf16)v.w;
        *(v4bf*)(sx + row * XSTR_ + c4 * 4) = w;
    }
    __syncthreads();

    v8f accG = {0.f,0.f,0.f,0.f,0.f,0.f,0.f,0.f};
    v8f accZ = {0.f,0.f,0.f,0.f,0.f,0.f,0.f,0.f};

    #pragma unroll
    for (int tap = 0; tap < 3; ++tap) {
        const __bf16* arow = sx + (size_t)(lr + tap) * XSTR_ + 8 * half;
        #pragma unroll
        for (int kk = 0; kk < 4; ++kk) {                 // D=128 in 4 chunks of 32
            // fragment id: K-chunk (tap*4+kk) of 12, n-tile = wave of 8
            size_t fo = ((size_t)((tap * 4 + kk) * 8 + wave) * 32 + lane) * 16;
            __builtin_prefetch(wgrep + fo + 8 * 32 * 16, 0, 0);
            v16bf afrag = frag_from_lds(arow + kk * 32);
            v16bf gfrag = *(const v16bf*)(wgrep + fo);
            v16bf zfrag = *(const v16bf*)(wzrep + fo);
            accG = __builtin_amdgcn_wmma_f32_16x16x32_bf16(false, afrag, false, gfrag,
                                                           (short)0, accG, false, false);
            accZ = __builtin_amdgcn_wmma_f32_16x16x32_bf16(false, afrag, false, zfrag,
                                                           (short)0, accZ, false, false);
        }
    }
    const float bG = biasG[n0 + lr];
    const float bZ = biasZ[n0 + lr];
    #pragma unroll
    for (int r = 0; r < 8; ++r) {
        int m = r + (half << 3);
        size_t o = ((size_t)b * T_ + t0 + m) * N_ + n0 + lr;
        float gv = accG[r] + bG; g[o] = gv > 0.f ? gv : 0.f;
        float zv = accZ[r] + bZ; z[o] = zv > 0.f ? zv : 0.f;
    }
}

// ---------------------------------------------------------------------------
// 3) Chunked parallel scan: h[t] = Abar[t]*h[t-1] + Bbar[t]*z[t]
//    pass1: per-chunk (prod A, local state) ; pass2: serial combine over 32
//    chunks ; pass3: replay chunk with true carry-in, y = C*h (gated on even t)
//    written in place over z.  Thread layout keeps lane index on n (coalesced).
// ---------------------------------------------------------------------------
__global__ void scan_pass1(const float* __restrict__ Abar, const float* __restrict__ Bbar,
                           const float* __restrict__ z,
                           float* __restrict__ chP, float* __restrict__ chS) {
    int i = blockIdx.x * blockDim.x + threadIdx.x;       // c*B*N + b*N + n
    if (i >= NC_ * B_ * N_) return;
    int n = i & (N_ - 1);
    int b = (i >> 7) & (B_ - 1);
    int c = i >> 10;
    float P = 1.f, S = 0.f;
    const float* zp = z + ((size_t)b * T_) * N_ + n;
    int t0 = c * CHUNK_;
    for (int j = 0; j < CHUNK_; ++j) {
        size_t ci = (size_t)(t0 + j) * N_ + n;
        float a = Abar[ci];
        S = a * S + Bbar[ci] * zp[(size_t)(t0 + j) * N_];
        P = a * P;
    }
    chP[i] = P; chS[i] = S;
}

__global__ void scan_pass2(const float* __restrict__ chP, const float* __restrict__ chS,
                           float* __restrict__ chH) {
    int i = blockIdx.x * blockDim.x + threadIdx.x;       // b*N + n
    if (i >= B_ * N_) return;
    float h = 0.f;
    for (int c = 0; c < NC_; ++c) {
        size_t o = (size_t)c * B_ * N_ + i;
        chH[o] = h;                                      // carry-in for chunk c
        h = chP[o] * h + chS[o];
    }
}

__global__ void scan_pass3(const float* __restrict__ Abar, const float* __restrict__ Bbar,
                           const float* __restrict__ Cc, const float* __restrict__ gate,
                           float* __restrict__ zy, const float* __restrict__ chH) {
    int i = blockIdx.x * blockDim.x + threadIdx.x;
    if (i >= NC_ * B_ * N_) return;
    int n = i & (N_ - 1);
    int b = (i >> 7) & (B_ - 1);
    int c = i >> 10;
    float h = chH[(size_t)c * B_ * N_ + b * N_ + n];
    int t0 = c * CHUNK_;
    for (int j = 0; j < CHUNK_; ++j) {
        int t = t0 + j;
        size_t ci = (size_t)t * N_ + n;
        size_t zi = ((size_t)b * T_ + t) * N_ + n;
        h = Abar[ci] * h + Bbar[ci] * zy[zi];
        float y = Cc[ci] * h;
        if (!(t & 1)) {                                  // spec tokens: gated
            float gv = gate[zi];
            y *= 1.f / (1.f + __expf(-gv));
        }
        zy[zi] = y;                                      // in-place over z
    }
}

// ---------------------------------------------------------------------------
// 4) Generic WMMA GEMM: C[M,Nn] = act(A[M,K] @ W[K,Nn] + bias), W repacked.
//    Block = 256 threads = 8 waves: one 16-row m-tile, waves span 8 n-tiles
//    (grid.y covers Nn > 128).  A tile staged once in LDS as bf16; rowMode
//    token-gather (coarse = even t, fine = odd t) applied at staging time.
//    act: 0 none, 1 relu, 2 sigmoid.
// ---------------------------------------------------------------------------
__global__ __launch_bounds__(256)
void gemm_kernel(const float* __restrict__ A, const __bf16* __restrict__ Wrep,
                 const float* __restrict__ bias, float* __restrict__ C,
                 int M, int Kdim, int Nn, int rowMode, int act) {
    __shared__ __bf16 sa[16 * ASTR_];

    const int lane = threadIdx.x & 31;
    const int wave = threadIdx.x >> 5;
    const int half = lane >> 4;
    const int lr   = lane & 15;
    const int m0   = blockIdx.x * 16;
    const int ntI  = blockIdx.y * 8 + wave;              // n-tile index
    const int n0   = ntI * 16;
    const int ntiles = Nn >> 4;

    // ---- cooperative stage: 16 rows x (Kdim/4) float4 chunks ----
    const int KW = Kdim >> 2;
    for (int q = threadIdx.x; q < 16 * KW; q += 256) {
        int row = q / KW;
        int c4  = q - row * KW;
        int m   = m0 + row;
        size_t abase;
        if (rowMode == 0)
            abase = (size_t)m * Kdim;
        else
            abase = ((size_t)(m >> 11) * T_ + 2 * (m & 2047) + (rowMode - 1)) * (size_t)D_;
        float4 v = *(const float4*)(A + abase + c4 * 4);
        v4bf w;
        w[0] = (__bf16)v.x; w[1] = (__bf16)v.y; w[2] = (__bf16)v.z; w[3] = (__bf16)v.w;
        *(v4bf*)(sa + row * ASTR_ + c4 * 4) = w;
    }
    __syncthreads();

    v8f acc = {0.f,0.f,0.f,0.f,0.f,0.f,0.f,0.f};

    const __bf16* arow = sa + (size_t)lr * ASTR_ + 8 * half;
    const int KT = Kdim >> 5;
    for (int kk = 0; kk < KT; ++kk) {
        size_t fo = ((size_t)(kk * ntiles + ntI) * 32 + lane) * 16;
        __builtin_prefetch(Wrep + fo + (size_t)ntiles * 32 * 16, 0, 0);
        v16bf afrag = frag_from_lds(arow + kk * 32);
        v16bf wfrag = *(const v16bf*)(Wrep + fo);
        acc = __builtin_amdgcn_wmma_f32_16x16x32_bf16(false, afrag, false, wfrag,
                                                      (short)0, acc, false, false);
    }

    const float bv = bias[n0 + lr];
    #pragma unroll
    for (int r = 0; r < 8; ++r) {
        int m = m0 + r + (half << 3);
        float v = acc[r] + bv;
        if (act == 1)      v = v > 0.f ? v : 0.f;
        else if (act == 2) v = 1.f / (1.f + __expf(-v));
        C[(size_t)m * Nn + n0 + lr] = v;
    }
}

// ---------------------------------------------------------------------------
// Host-side orchestration
// ---------------------------------------------------------------------------
static inline void launch_repack(const float* W, __bf16* out, int Kdim, int Nn,
                                 hipStream_t stream) {
    int total = (Kdim / 32) * (Nn / 16) * 32;
    repack_w_kernel<<<(total + 255) / 256, 256, 0, stream>>>(W, out, Kdim, Nn);
}

extern "C" void kernel_launch(void* const* d_in, const int* in_sizes, int n_in,
                              void* d_out, int out_size, void* d_ws, size_t ws_size,
                              hipStream_t stream) {
    (void)in_sizes; (void)n_in; (void)out_size; (void)ws_size;

    const float* x_in = (const float*)d_in[0];
    // d_in[1] = type_ids (fixed t%2 pattern, folded into kernels statically)

    float* ws = (float*)d_ws;
    const size_t SZ_BTN = (size_t)B_ * T_ * N_;          // 4,194,304
    const size_t SZ_TN  = (size_t)T_ * N_;               //   524,288
    float* xA   = ws;                                    // block activations ping
    float* xB   = xA   + SZ_BTN;                         // block activations pong
    float* gbuf = xB   + SZ_BTN;                         // gate (pre-sigmoid)
    float* zbuf = gbuf + SZ_BTN;                         // z, overwritten by y
    float* Abar = zbuf + SZ_BTN;
    float* Bbar = Abar + SZ_TN;
    float* Cc   = Bbar + SZ_TN;
    float* chP  = Cc   + SZ_TN;                          // NC*B*N = 32768 each
    float* chS  = chP  + (size_t)NC_ * B_ * N_;
    float* chH  = chS  + (size_t)NC_ * B_ * N_;
    float* hbuf = chH  + (size_t)NC_ * B_ * N_;          // head hidden [16384,256]
    // bf16 repacked-weight region (32 B aligned: float offset * 4)
    __bf16* wrep0 = (__bf16*)(hbuf + (size_t)TOK_ * HID_);   // conv wg / head mats
    __bf16* wrep1 = wrep0 + 3 * D_ * N_;                     // conv wz / head mats

    const float* cur = x_in;
    for (int blk = 0; blk < NBLK_; ++blk) {
        const float* const* bp = (const float* const*)(d_in + 2 + blk * 13);
        const float *wg = bp[0], *bg = bp[1], *wz = bp[2], *bz = bp[3];
        const float *A1 = bp[4], *A2 = bp[5], *B1 = bp[6], *B2 = bp[7];
        const float *C1 = bp[8], *C2 = bp[9], *dl = bp[10], *wo = bp[11], *bo = bp[12];
        float* nxt = (blk & 1) ? xB : xA;

        // conv weights are [3,D,N] = contiguous [384,128] row-major
        launch_repack(wg, wrep0, 3 * D_, N_, stream);
        launch_repack(wz, wrep1, 3 * D_, N_, stream);
        coef_kernel<<<(T_ * N_) / 256, 256, 0, stream>>>(A1, A2, B1, B2, C1, C2, dl,
                                                         Abar, Bbar, Cc);
        conv_gz_kernel<<<dim3(T_ / 16, B_), 256, 0, stream>>>(cur, wrep0, bg, wrep1, bz,
                                                              gbuf, zbuf);
        scan_pass1<<<(NC_ * B_ * N_) / 256, 256, 0, stream>>>(Abar, Bbar, zbuf, chP, chS);
        scan_pass2<<<(B_ * N_) / 256, 256, 0, stream>>>(chP, chS, chH);
        scan_pass3<<<(NC_ * B_ * N_) / 256, 256, 0, stream>>>(Abar, Bbar, Cc, gbuf,
                                                              zbuf, chH);
        // x_next = y @ wo + bo   (M = B*T, K = N = 128)
        launch_repack(wo, wrep0, N_, D_, stream);
        gemm_kernel<<<dim3((B_ * T_) / 16, 1), 256, 0, stream>>>(zbuf, wrep0, bo, nxt,
                                                                 B_ * T_, N_, D_, 0, 0);
        cur = nxt;
    }

    // Heads: hd 0 = coarse (even t), hd 1 = first/fine (odd t)
    float* outp = (float*)d_out;
    for (int hd = 0; hd < 2; ++hd) {
        const float* const* hp = (const float* const*)(d_in + 2 + NBLK_ * 13 + hd * 10);
        const float *w = hp[0], *b = hp[1];
        const float *wv = hp[2], *bv = hp[3], *wf = hp[4], *bf = hp[5];
        const float *won = hp[6], *bon = hp[7], *woff = hp[8], *boff = hp[9];
        const int rowMode = hd + 1;
        // hidden = tok @ w + b   (M = 16384, K = 128, N = 256), token gather fused
        launch_repack(w, wrep0, D_, HID_, stream);
        gemm_kernel<<<dim3(TOK_ / 16, HID_ / 128), 256, 0, stream>>>(cur, wrep0, b, hbuf,
                                                                     TOK_, D_, HID_,
                                                                     rowMode, 0);
        float* base = outp + (size_t)hd * 4 * TOKP_;
        const float* pw[4] = {wv, wf, won, woff};
        const float* pb[4] = {bv, bf, bon, boff};
        const int    pact[4] = {0, 2, 2, 2};             // velocity linear, rest sigmoid
        for (int o = 0; o < 4; ++o) {
            launch_repack(pw[o], wrep1, HID_, P_, stream);
            gemm_kernel<<<dim3(TOK_ / 16, 1), 256, 0, stream>>>(hbuf, wrep1, pb[o],
                                                                base + (size_t)o * TOKP_,
                                                                TOK_, HID_, P_, 0,
                                                                pact[o]);
        }
    }
}
